// DMPNN_32727650795770
// MI455X (gfx1250) — matrix operand order repaired and checked by reading, not compiled, not run on hardware
//
#include <hip/hip_runtime.h>
#include <math.h>

#define NN   20000
#define EE   120000
#define GG   50
#define NPG  400     // NN / GG
#define DEG  6
#define MT   64      // M rows per GEMM block (4 WMMA row-subtiles)

typedef __attribute__((ext_vector_type(16))) _Float16 v16h;
typedef __attribute__((ext_vector_type(8)))  _Float16 v8h;
typedef __attribute__((ext_vector_type(8)))  float    v8f;

// ---------------------------------------------------------------------------
// Pack fp32 weights W[K,Nout] into this file's WMMA f16 B-operand order (contiguous k):
// Wp[((kc*nnt + nt)*32 + lane)*16 + j] = W[kc*32 + (lane<16?0:16) + j][nt*16 + (lane&15)]
// (lanes 0-15 hold k 0-15, lanes 16-31 hold k 16-31; the A loads use the same order)
// ---------------------------------------------------------------------------
__global__ void pack_weights(const float* __restrict__ W, _Float16* __restrict__ Wp,
                             int K, int Nout) {
  int nkt = (K + 31) >> 5;
  int nnt = Nout >> 4;
  int total = nkt * nnt * 512;
  int idx = blockIdx.x * blockDim.x + threadIdx.x;
  if (idx >= total) return;
  int j    = idx & 15;
  int lane = (idx >> 4) & 31;
  int rest = idx >> 9;
  int nt   = rest % nnt;
  int kc   = rest / nnt;
  int k = kc * 32 + ((lane < 16) ? 0 : 16) + j;
  int n = nt * 16 + (lane & 15);
  Wp[idx] = (_Float16)((k < K) ? W[(long)k * Nout + n] : 0.0f);
}

// ---------------------------------------------------------------------------
// Fused WMMA GEMM: out = act( Aload[M,K] @ W[K,Nout] + bias + resid )
// Block = 64 output rows; one wave per 16-column tile (blockDim = 2*Nout).
// Per K-chunk: one B register load reused by 4 WMMAs (4 row-subtiles).
// A-loader MODE (compile-time):
//   0: plain A[M,K]                     (K must be even for the fast path)
//   1: row r = concat( hnode[src[r]][0:128], hef[r][0:64], dist[r] )   (K=193)
//   2: row r = concat( A[r][0:128], A1[r][0:128] )                     (K=256)
// ---------------------------------------------------------------------------
template <int MODE>
__global__ void wmma_gemm_t(const float* __restrict__ A,
                            const float* __restrict__ A1,
                            const float* __restrict__ A2,
                            const int*   __restrict__ srcIdx,
                            const _Float16* __restrict__ Wp,
                            const float* __restrict__ bias,
                            const float* __restrict__ resid,
                            float* __restrict__ out,
                            int M, int K, int Nout, int act) {
  const int nkt  = (K + 31) >> 5;
  const int nnt  = Nout >> 4;
  const int row0 = blockIdx.x * MT;
  const int tid  = threadIdx.x;
  const int lane = tid & 31;
  const int wv   = tid >> 5;
  __shared__ _Float16 As[MT * 32];

  v8f acc[4] = {};
  const int am = lane & 15;
  const int kb = (lane < 16) ? 0 : 16;
  const bool fullM = (row0 + MT) <= M;

  for (int kc = 0; kc < nkt; ++kc) {
    const bool fullK = ((kc << 5) + 32) <= K;
    if (fullM && fullK) {
      // ---- fast path: uniform, vectorized, 8 elements / thread / group ----
      for (int base = tid * 8; base < MT * 32; base += blockDim.x * 8) {
        int r = base >> 5, kk = base & 31;
        int gr = row0 + r, gk = (kc << 5) + kk;
        float f[8];
        if (MODE == 0) {
          const float* p = &A[(long)gr * K + gk];        // 8B aligned (K even)
          float2 a0 = *(const float2*)(p + 0);
          float2 a1 = *(const float2*)(p + 2);
          float2 a2 = *(const float2*)(p + 4);
          float2 a3 = *(const float2*)(p + 6);
          f[0] = a0.x; f[1] = a0.y; f[2] = a1.x; f[3] = a1.y;
          f[4] = a2.x; f[5] = a2.y; f[6] = a3.x; f[7] = a3.y;
        } else if (MODE == 1) {                          // gk < 192 guaranteed
          const float* p = (gk < 128) ? &A[(long)srcIdx[gr] * 128 + gk]
                                      : &A1[(long)gr * 64 + (gk - 128)];
          float4 a0 = *(const float4*)(p + 0);           // rows 16B aligned
          float4 a1 = *(const float4*)(p + 4);
          f[0] = a0.x; f[1] = a0.y; f[2] = a0.z; f[3] = a0.w;
          f[4] = a1.x; f[5] = a1.y; f[6] = a1.z; f[7] = a1.w;
        } else {                                         // MODE 2, rows of 128
          const float* p = (gk < 128) ? &A[(long)gr * 128 + gk]
                                      : &A1[(long)gr * 128 + (gk - 128)];
          float4 a0 = *(const float4*)(p + 0);
          float4 a1 = *(const float4*)(p + 4);
          f[0] = a0.x; f[1] = a0.y; f[2] = a0.z; f[3] = a0.w;
          f[4] = a1.x; f[5] = a1.y; f[6] = a1.z; f[7] = a1.w;
        }
        v8h hv;
#pragma unroll
        for (int i = 0; i < 8; ++i) hv[i] = (_Float16)f[i];
        *(v8h*)(&As[base]) = hv;                         // one ds_store_b128
      }
    } else {
      // ---- slow path: boundary tiles / partial K chunks ----
      for (int idx = tid; idx < MT * 32; idx += blockDim.x) {
        int r = idx >> 5, kk = idx & 31;
        int gr = row0 + r, gk = (kc << 5) + kk;
        float v = 0.0f;
        if (gr < M && gk < K) {
          if (MODE == 0) {
            v = A[(long)gr * K + gk];
          } else if (MODE == 1) {
            if (gk < 128)      v = A[(long)srcIdx[gr] * 128 + gk];
            else if (gk < 192) v = A1[(long)gr * 64 + (gk - 128)];
            else               v = A2[gr];               // gk == 192: dist
          } else {
            v = (gk < 128) ? A[(long)gr * 128 + gk]
                           : A1[(long)gr * 128 + (gk - 128)];
          }
        }
        As[idx] = (_Float16)v;
      }
    }
    __syncthreads();
    // B operand: contiguous 32B per lane, reused across 4 row-subtiles
    v16h b = *(const v16h*)(Wp + (((long)(kc * nnt + wv) * 32 + lane) << 4));
#pragma unroll
    for (int s = 0; s < 4; ++s) {
      // A operand (contiguous k, same order as packed B): lane m / m+16 split K halves
      const _Float16* ap = &As[(s * 16 + am) * 32 + kb];
      v16h a;
#pragma unroll
      for (int i = 0; i < 8; ++i) a[i] = ap[i];           // K = kb .. kb+7
#pragma unroll
      for (int i = 0; i < 8; ++i) a[8 + i] = ap[8 + i];   // K = kb+8 .. kb+15
      acc[s] = __builtin_amdgcn_wmma_f32_16x16x32_f16(false, a, false, b,
                                                      (short)0, acc[s],
                                                      false, false);
    }
    __syncthreads();
  }
  // epilogue: C/D layout — VGPR v holds row v (lanes 0-15) / v+8 (lanes 16-31)
  const int n  = (wv << 4) + (lane & 15);
  const int mb = (lane < 16) ? 0 : 8;
  const float bb = bias ? bias[n] : 0.0f;
#pragma unroll
  for (int s = 0; s < 4; ++s) {
#pragma unroll
    for (int v = 0; v < 8; ++v) {
      int gr = row0 + s * 16 + mb + v;
      if (gr < M) {
        float x = acc[s][v] + bb;
        if (resid) x += resid[(long)gr * Nout + n];
        if (act) x = x > 0.0f ? x : 0.0f;
        out[(long)gr * Nout + n] = x;
      }
    }
  }
}

// ---------------------------------------------------------------------------
// Per-edge geometry: dist and dd (dd is the att output -> written to d_out)
// ---------------------------------------------------------------------------
__global__ void edge_geom(const float* __restrict__ coord, const float* __restrict__ aw,
                          const int* __restrict__ src, const int* __restrict__ dst,
                          float* __restrict__ dist, float* __restrict__ dd) {
  int e = blockIdx.x * blockDim.x + threadIdx.x;
  if (e >= EE) return;
  int s = src[e], t = dst[e];
  float dx = coord[3 * s]     - coord[3 * t];
  float dy = coord[3 * s + 1] - coord[3 * t + 1];
  float dz = coord[3 * s + 2] - coord[3 * t + 2];
  dist[e] = sqrtf(dx * dx + dy * dy + dz * dz);
  float a = aw[e];
  dd[e] = (a == 1.0f) ? 1.0f : ((a == 0.0f) ? -1.0f : logf(a) * 2.0f);
}

// ---------------------------------------------------------------------------
// Directed message: m[e] = sum over incoming edges j of src[e]
//   (j = 6*src[e]..+5, contiguous) of h[j]*dd[j], masked where src[j]==dst[e].
// h (61 MB) is L2-resident, so the 6-way gather hits L2.
// ---------------------------------------------------------------------------
__global__ void msg_kernel(const float* __restrict__ h, const float* __restrict__ dd,
                           const int* __restrict__ src, float* __restrict__ m) {
  int e = blockIdx.x;
  int c = threadIdx.x;              // 128 columns
  int s = src[e];
  int t = e / DEG;                  // dst[e] by construction
  __shared__ int   ssrc[DEG];
  __shared__ float sdd[DEG];
  if (c < DEG) { int j = DEG * s + c; ssrc[c] = src[j]; sdd[c] = dd[j]; }
  __syncthreads();
  float acc = 0.0f;
#pragma unroll
  for (int k = 0; k < DEG; ++k) {
    if (ssrc[k] != t) acc += h[(long)(DEG * s + k) * 128 + c] * sdd[k];
  }
  m[(long)e * 128 + c] = acc;
}

// agg[v] = sum of h over the contiguous incoming-edge block of v
__global__ void agg_kernel(const float* __restrict__ h, float* __restrict__ agg) {
  int idx = blockIdx.x * blockDim.x + threadIdx.x;
  if (idx >= NN * 128) return;
  int v = idx >> 7, c = idx & 127;
  float acc = 0.0f;
#pragma unroll
  for (int k = 0; k < DEG; ++k) acc += h[(long)(DEG * v + k) * 128 + c];
  agg[idx] = acc;
}

// g[gi] = sum of hn over the contiguous 400-node block of graph gi
__global__ void g_init(const float* __restrict__ hn, float* __restrict__ g) {
  int gi = blockIdx.x, c = threadIdx.x;  // 128
  float acc = 0.0f;
  long base = (long)gi * NPG;
  for (int n = 0; n < NPG; ++n) acc += hn[(base + n) * 128 + c];
  g[gi * 128 + c] = acc;
}

// ---------------------------------------------------------------------------
// Per-graph attention softmax + context (block per graph, 256 threads)
// ---------------------------------------------------------------------------
__global__ void attn_ctx(const float* __restrict__ hn, const float* __restrict__ proj,
                         const float* __restrict__ g, const float* __restrict__ W_att,
                         const float* __restrict__ b_att, float* __restrict__ ctx) {
  int gi = blockIdx.x, tid = threadIdx.x;
  __shared__ float lg[NPG];
  __shared__ float red[256];
  __shared__ float sWh[128];
  if (tid < 128) sWh[tid] = W_att[128 + tid];
  // gdot = dot(g[gi], W_att[0:128])
  float p = (tid < 128) ? g[gi * 128 + tid] * W_att[tid] : 0.0f;
  red[tid] = p; __syncthreads();
  for (int s = 128; s > 0; s >>= 1) { if (tid < s) red[tid] += red[tid + s]; __syncthreads(); }
  float gdot = red[0] + b_att[0];
  __syncthreads();
  // logits
  for (int n = tid; n < NPG; n += 256) {
    long node = (long)gi * NPG + n;
    float hd = 0.0f;
    for (int c = 0; c < 128; ++c) hd += hn[node * 128 + c] * sWh[c];
    float l = gdot + hd;
    lg[n] = (l > 0.0f) ? l : 0.01f * l;   // leaky_relu
  }
  __syncthreads();
  // softmax: max
  float lm = -1e30f;
  for (int n = tid; n < NPG; n += 256) lm = fmaxf(lm, lg[n]);
  red[tid] = lm; __syncthreads();
  for (int s = 128; s > 0; s >>= 1) { if (tid < s) red[tid] = fmaxf(red[tid], red[tid + s]); __syncthreads(); }
  float mx = red[0]; __syncthreads();
  // exp + sum
  float ls = 0.0f;
  for (int n = tid; n < NPG; n += 256) { float ev = expf(lg[n] - mx); lg[n] = ev; ls += ev; }
  red[tid] = ls; __syncthreads();
  for (int s = 128; s > 0; s >>= 1) { if (tid < s) red[tid] += red[tid + s]; __syncthreads(); }
  float inv = 1.0f / red[0]; __syncthreads();
  // ctx[c] = elu( sum_n a[n]*proj[n][c] )
  if (tid < 128) {
    float acc = 0.0f;
    long base = (long)gi * NPG;
    for (int n = 0; n < NPG; ++n) acc += lg[n] * inv * proj[(base + n) * 128 + tid];
    ctx[gi * 128 + tid] = (acc > 0.0f) ? acc : expm1f(acc);
  }
}

// GRU update (block per graph, 384 threads); g updated in place
__global__ void gru_kernel(const float* __restrict__ ctx, float* __restrict__ g,
                           const float* __restrict__ Wg, const float* __restrict__ Ug,
                           const float* __restrict__ bg) {
  int gi = blockIdx.x, j = threadIdx.x;  // 384
  __shared__ float sgx[384], sgh[384], sctx[128], sg[128];
  if (j < 128) { sctx[j] = ctx[gi * 128 + j]; sg[j] = g[gi * 128 + j]; }
  __syncthreads();
  float gx = bg[j], gh = 0.0f;
  for (int c = 0; c < 128; ++c) {
    gx += sctx[c] * Wg[c * 384 + j];
    gh += sg[c]   * Ug[c * 384 + j];
  }
  sgx[j] = gx; sgh[j] = gh; __syncthreads();
  if (j < 128) {
    float r = 1.0f / (1.0f + expf(-(sgx[j] + sgh[j])));
    float z = 1.0f / (1.0f + expf(-(sgx[128 + j] + sgh[128 + j])));
    float n = tanhf(sgx[256 + j] + r * sgh[256 + j]);
    g[gi * 128 + j] = (1.0f - z) * n + z * sg[j];
  }
}

// MLP head (block per graph, 256 threads)
__global__ void head_kernel(const float* __restrict__ hs1, const float* __restrict__ hs2,
                            const float* __restrict__ W1, const float* __restrict__ b1,
                            const float* __restrict__ W2, const float* __restrict__ b2,
                            const float* __restrict__ W3, const float* __restrict__ b3,
                            const float* __restrict__ W4, const float* __restrict__ b4,
                            float* __restrict__ zk) {
  int gi = blockIdx.x, tid = threadIdx.x;
  __shared__ float x0[384], x1[256], x2[128], x3[64];
  if (tid < 128) {
    float a = hs1[gi * 128 + tid], b = hs2[gi * 128 + tid];
    x0[tid] = a; x0[128 + tid] = b; x0[256 + tid] = a - b;
  }
  __syncthreads();
  if (tid < 256) { float acc = b1[tid]; for (int c = 0; c < 384; ++c) acc += x0[c] * W1[c * 256 + tid];
                   x1[tid] = acc > 0.0f ? acc : 0.0f; }
  __syncthreads();
  if (tid < 128) { float acc = b2[tid]; for (int c = 0; c < 256; ++c) acc += x1[c] * W2[c * 128 + tid];
                   x2[tid] = acc > 0.0f ? acc : 0.0f; }
  __syncthreads();
  if (tid < 64)  { float acc = b3[tid]; for (int c = 0; c < 128; ++c) acc += x2[c] * W3[c * 64 + tid];
                   x3[tid] = acc > 0.0f ? acc : 0.0f; }
  __syncthreads();
  if (tid == 0)  { float acc = b4[0];   for (int c = 0; c < 64; ++c)  acc += x3[c] * W4[c];
                   zk[gi] = acc; }
}

// ---------------------------------------------------------------------------
extern "C" void kernel_launch(void* const* d_in, const int* in_sizes, int n_in,
                              void* d_out, int out_size, void* d_ws, size_t ws_size,
                              hipStream_t stream) {
  (void)in_sizes; (void)n_in; (void)out_size; (void)ws_size;
  const float* atom1  = (const float*)d_in[0];
  const float* coord1 = (const float*)d_in[1];
  const float* efeat1 = (const float*)d_in[2];
  const float* aw1    = (const float*)d_in[3];
  const float* atom2  = (const float*)d_in[4];
  const float* coord2 = (const float*)d_in[5];
  const float* efeat2 = (const float*)d_in[6];
  const float* aw2    = (const float*)d_in[7];
  const float* W_atom = (const float*)d_in[8];   const float* b_atom = (const float*)d_in[9];
  const float* W_edge = (const float*)d_in[10];  const float* b_edge = (const float*)d_in[11];
  const float* W_init = (const float*)d_in[12];  const float* b_init = (const float*)d_in[13];
  const float* W_msg  = (const float*)d_in[14];  const float* b_msg  = (const float*)d_in[15];
  const float* W_node = (const float*)d_in[16];  const float* b_node = (const float*)d_in[17];
  const float* W_att  = (const float*)d_in[18];  const float* b_att  = (const float*)d_in[19];
  const float* W_proj = (const float*)d_in[20];  const float* b_proj = (const float*)d_in[21];
  const float* Wg     = (const float*)d_in[22];  const float* Ug     = (const float*)d_in[23];
  const float* bg     = (const float*)d_in[24];
  const float* W1 = (const float*)d_in[25]; const float* b1 = (const float*)d_in[26];
  const float* W2 = (const float*)d_in[27]; const float* b2 = (const float*)d_in[28];
  const float* W3 = (const float*)d_in[29]; const float* b3 = (const float*)d_in[30];
  const float* W4 = (const float*)d_in[31]; const float* b4 = (const float*)d_in[32];
  const int* src1 = (const int*)d_in[33]; const int* dst1 = (const int*)d_in[34];
  const int* src2 = (const int*)d_in[38]; const int* dst2 = (const int*)d_in[39];
  float* dout = (float*)d_out;  // [zk (50) | att1 (E) | att2 (E)]

  // ---- workspace carving ----
  size_t wsOff = 0;
  auto WS = [&](size_t nbytes) -> void* {
    void* p = (char*)d_ws + wsOff;
    wsOff += (nbytes + 255) & ~(size_t)255;
    return p;
  };
  _Float16* WpAtom = (_Float16*)WS((size_t)3 * 8 * 512 * 2);  // K=70  -> 3 chunks
  _Float16* WpEdge = (_Float16*)WS((size_t)1 * 4 * 512 * 2);  // K=14, Nout=64
  _Float16* WpInit = (_Float16*)WS((size_t)7 * 8 * 512 * 2);  // K=193 -> 7 chunks
  _Float16* WpMsg  = (_Float16*)WS((size_t)4 * 8 * 512 * 2);  // K=128
  _Float16* WpNode = (_Float16*)WS((size_t)8 * 8 * 512 * 2);  // K=256
  _Float16* WpProj = (_Float16*)WS((size_t)4 * 8 * 512 * 2);  // K=128
  float* hnode = (float*)WS((size_t)NN * 128 * 4);
  float* hef   = (float*)WS((size_t)EE * 64 * 4);
  float* dist  = (float*)WS((size_t)EE * 4);
  float* h0    = (float*)WS((size_t)EE * 128 * 4);
  float* hbuf  = (float*)WS((size_t)EE * 128 * 4);
  float* mbuf  = (float*)WS((size_t)EE * 128 * 4);
  float* aggB  = (float*)WS((size_t)NN * 128 * 4);
  float* hn    = (float*)WS((size_t)NN * 128 * 4);
  float* projB = (float*)WS((size_t)NN * 128 * 4);
  float* g     = (float*)WS((size_t)GG * 128 * 4);
  float* ctx   = (float*)WS((size_t)GG * 128 * 4);
  float* hs1   = (float*)WS((size_t)GG * 128 * 4);
  float* hs2   = (float*)WS((size_t)GG * 128 * 4);

  // ---- pack weights into WMMA B layout (once per call) ----
  auto pack = [&](const float* W, _Float16* Wp, int K, int Nout) {
    int total = ((K + 31) >> 5) * (Nout >> 4) * 512;
    pack_weights<<<dim3((total + 255) / 256), dim3(256), 0, stream>>>(W, Wp, K, Nout);
  };
  pack(W_atom, WpAtom, 70, 128);
  pack(W_edge, WpEdge, 14, 64);
  pack(W_init, WpInit, 193, 128);
  pack(W_msg,  WpMsg,  128, 128);
  pack(W_node, WpNode, 256, 128);
  pack(W_proj, WpProj, 128, 128);

  const dim3 gN((NN + MT - 1) / MT), gE((EE + MT - 1) / MT);
  const dim3 t256(256), t128(128);

  auto runGraph = [&](const float* atom, const float* coord, const float* efeat,
                      const float* aw, const int* src, const int* dst,
                      float* ddout, float* hs) {
    // node embedding: relu(atom @ W_atom + b)
    wmma_gemm_t<0><<<gN, t256, 0, stream>>>(atom, nullptr, nullptr, nullptr,
        WpAtom, b_atom, nullptr, hnode, NN, 70, 128, 1);
    // edge embedding: relu(efeat @ W_edge + b)
    wmma_gemm_t<0><<<gE, t128, 0, stream>>>(efeat, nullptr, nullptr, nullptr,
        WpEdge, b_edge, nullptr, hef, EE, 14, 64, 1);
    edge_geom<<<dim3((EE + 255) / 256), t256, 0, stream>>>(coord, aw, src, dst, dist, ddout);
    // h0 = relu(concat(hnode[src], hef, dist) @ W_init + b)
    wmma_gemm_t<1><<<gE, t256, 0, stream>>>(hnode, hef, dist, src,
        WpInit, b_init, nullptr, h0, EE, 193, 128, 1);
    // RADIUS=3 directed message passing
    const float* hcur = h0;
    for (int r = 0; r < 3; ++r) {
      msg_kernel<<<dim3(EE), t128, 0, stream>>>(hcur, ddout, src, mbuf);
      wmma_gemm_t<0><<<gE, t256, 0, stream>>>(mbuf, nullptr, nullptr, nullptr,
          WpMsg, b_msg, h0, hbuf, EE, 128, 128, 1);  // relu(h0 + m@W_msg + b)
      hcur = hbuf;
    }
    agg_kernel<<<dim3((NN * 128 + 255) / 256), t256, 0, stream>>>(hbuf, aggB);
    // hn = relu(concat(hnode, agg) @ W_node + b)
    wmma_gemm_t<2><<<gN, t256, 0, stream>>>(hnode, aggB, nullptr, nullptr,
        WpNode, b_node, nullptr, hn, NN, 256, 128, 1);
    // readout
    g_init<<<dim3(GG), t128, 0, stream>>>(hn, g);
    wmma_gemm_t<0><<<gN, t256, 0, stream>>>(hn, nullptr, nullptr, nullptr,
        WpProj, b_proj, nullptr, projB, NN, 128, 128, 0);  // no activation
    for (int t = 0; t < 2; ++t) {
      attn_ctx<<<dim3(GG), t256, 0, stream>>>(hn, projB, g, W_att, b_att, ctx);
      gru_kernel<<<dim3(GG), dim3(384), 0, stream>>>(ctx, g, Wg, Ug, bg);
    }
    hipMemcpyAsync(hs, g, (size_t)GG * 128 * sizeof(float),
                   hipMemcpyDeviceToDevice, stream);
  };

  runGraph(atom1, coord1, efeat1, aw1, src1, dst1, dout + GG,       hs1);
  runGraph(atom2, coord2, efeat2, aw2, src2, dst2, dout + GG + EE,  hs2);

  head_kernel<<<dim3(GG), t256, 0, stream>>>(hs1, hs2, W1, b1, W2, b2, W3, b3, W4, b4, dout);
}
